// SingleNet_19808389169762
// MI455X (gfx1250) — matrix operand order repaired and checked by reading, not compiled, not run on hardware
//
#include <hip/hip_runtime.h>

// CDNA5 / gfx1250 WMMA fragment types (probe-confirmed signatures)
typedef __attribute__((ext_vector_type(16))) __bf16 v16bf;
typedef __attribute__((ext_vector_type(8)))  float  v8f;

#if __has_builtin(__builtin_amdgcn_tensor_load_to_lds) && \
    __has_builtin(__builtin_amdgcn_s_wait_tensorcnt)
#define USE_TDM 1
#else
#define USE_TDM 0
#endif

// ---------------------------------------------------------------------------
// Main MLP layer: out[32,O] = relu(A[32,K] @ W[O,K]^T + bias)
// One wave computes one 16x16 output tile via v_wmma_f32_16x16x32_bf16.
// Fragment layouts used here (contiguous k order, same for A and B):
//   A (16x32 bf16): lane l -> row M=l%16; elem e -> K = e + 16*(l/16)
//   B (32x16 bf16): lane l -> col N=l%16; elem e -> K = e + 16*(l/16)
//   C/D (16x16 f32): vgpr r, lane l -> (M = r + 8*(l/16), N = l%16)
// K and O are template constants so strides fold into immediate offsets
// (kills the 64-bit address-add chains seen in round-1 asm).
// ---------------------------------------------------------------------------
template <int K, int O>
__global__ __launch_bounds__(256) void mlp_layer_kernel(
    const float* __restrict__ A,     // [32, K] activations (fp32)
    const float* __restrict__ W,     // [O, K]  weights (fp32)
    const float* __restrict__ bias,  // [O]
    float* __restrict__ out)         // [32, O]
{
    const int lane = threadIdx.x & 31;
    const int wv   = threadIdx.x >> 5;
    const int lr   = lane & 15;      // N (and M-row) index within tile
    const int hi   = lane >> 4;      // 0 / 1
    const int nbase = (blockIdx.x * 8 + wv) * 16;
    const int mbase = blockIdx.y * 16;

    const float* Ar = A + (size_t)(mbase + lr) * K + 16 * hi;    // A row, K-pattern base
    const float* Wr = W + (size_t)(nbase + lr) * K + 16 * hi;   // W row, K-pattern base

    v8f c;
    {
        const float bv = bias[nbase + lr];         // bias depends on N only
        #pragma unroll
        for (int r = 0; r < 8; ++r) c[r] = bv;
    }

    #pragma unroll 2
    for (int k = 0; k < K; k += 32) {
        __builtin_prefetch(Wr + k + 256, 0, 0);    // global_prefetch_b8

        // ---- A fragment (16B-aligned float4 loads) ----
        const float4* ap0 = (const float4*)(Ar + k);
        const float4* ap1 = (const float4*)(Ar + k + 8);
        float4 a0 = ap0[0], a1 = ap0[1], a2 = ap1[0], a3 = ap1[1];
        v16bf af;
        af[0]=(__bf16)a0.x; af[1]=(__bf16)a0.y; af[2]=(__bf16)a0.z; af[3]=(__bf16)a0.w;
        af[4]=(__bf16)a1.x; af[5]=(__bf16)a1.y; af[6]=(__bf16)a1.z; af[7]=(__bf16)a1.w;
        af[8]=(__bf16)a2.x; af[9]=(__bf16)a2.y; af[10]=(__bf16)a2.z; af[11]=(__bf16)a2.w;
        af[12]=(__bf16)a3.x; af[13]=(__bf16)a3.y; af[14]=(__bf16)a3.z; af[15]=(__bf16)a3.w;

        // ---- B fragment: 16 consecutive K values of weight row nbase+lr ----
        const float4* bp = (const float4*)(Wr + k);
        float4 b0 = bp[0], b1 = bp[1], b2 = bp[2], b3 = bp[3];
        v16bf bf;
        bf[0]=(__bf16)b0.x; bf[1]=(__bf16)b0.y; bf[2]=(__bf16)b0.z; bf[3]=(__bf16)b0.w;
        bf[4]=(__bf16)b1.x; bf[5]=(__bf16)b1.y; bf[6]=(__bf16)b1.z; bf[7]=(__bf16)b1.w;
        bf[8]=(__bf16)b2.x; bf[9]=(__bf16)b2.y; bf[10]=(__bf16)b2.z; bf[11]=(__bf16)b2.w;
        bf[12]=(__bf16)b3.x; bf[13]=(__bf16)b3.y; bf[14]=(__bf16)b3.z; bf[15]=(__bf16)b3.w;

        c = __builtin_amdgcn_wmma_f32_16x16x32_bf16(
                false, af, false, bf, (short)0, c, false, false);
    }

    float* op = out + (size_t)(mbase + 8 * hi) * O + nbase + lr;
    #pragma unroll
    for (int r = 0; r < 8; ++r)
        op[(size_t)r * O] = fmaxf(c[r], 0.0f);     // constant-stride store clause
}

// ---------------------------------------------------------------------------
// Meta update: for every weight w_ij of W[O,I]:
//   mb2 + mw2 . relu(mw1 @ [v_in[i], w_ij, v_out[j]] + mb1)
// Layer-1 (K=3 padded to 32) runs on WMMA; relu(H) transposed through LDS;
// each lane then reduces one triple with mw2 held in registers (s_loads).
// Each block owns 1024 consecutive weights. The block's 4 KB weight chunk is
// streamed into LDS by the Tensor Data Mover (one TDM op per block, wave 0),
// synchronized with s_wait_tensorcnt — the CDNA5 async-tensor path.
// Control flow is wave-uniform, so EXEC is all-ones at every WMMA.
// ---------------------------------------------------------------------------
template <int ISHIFT>
__global__ __launch_bounds__(256) void meta_update_kernel(
    const float* __restrict__ W,      // [O, I] flat
    const float* __restrict__ vin,    // [I]  batch-0 input activations
    const float* __restrict__ vout,   // [O]  batch-0 output activations
    const float* __restrict__ mw1,    // [32, 3]
    const float* __restrict__ mb1,    // [32]
    const float* __restrict__ mw2,    // [1, 32]
    const float* __restrict__ mb2,    // [1]
    float* __restrict__ Wout)         // [O, I] flat
{
    __shared__ float Hb[8][32][36];   // stride 36: 16B-aligned rows, b128 reads
    __shared__ float Wbuf[1024];      // this block's weight chunk (TDM target)

    const int tid  = threadIdx.x;
    const int lane = tid & 31;
    const int wv   = tid >> 5;
    const int lr   = lane & 15;
    const int hi   = lane >> 4;
    const unsigned imask = (1u << ISHIFT) - 1u;
    const unsigned blkBase = blockIdx.x * 1024u;

#if USE_TDM
    if (wv == 0) {
        typedef unsigned int u32x4 __attribute__((ext_vector_type(4)));
        typedef int          i32x4 __attribute__((ext_vector_type(4)));
        typedef int          i32x8 __attribute__((ext_vector_type(8)));
        const unsigned long long ga = (unsigned long long)(size_t)(W + blkBase);
        const unsigned ldsb = (unsigned)(size_t)&Wbuf[0];   // LDS byte offset
        // D# group 0: count=1 | lds_addr | global_addr(57b) | type=2
        u32x4 g0;
        g0[0] = 1u;
        g0[1] = ldsb;
        g0[2] = (unsigned)ga;
        g0[3] = ((unsigned)(ga >> 32) & 0x01FFFFFFu) | 0x80000000u;
        // D# group 1: data_size=4B; tensor_dim0=1024, tensor_dim1=1;
        //             tile_dim0=1024, tile_dim1=1; dim0_stride=1024
        i32x8 g1;
        g1[0] = 0x00020000;            // [17:16] data_size = 2 (4 bytes)
        g1[1] = (int)(1024u << 16);    // tensor_dim0[15:0] in bits [63:48]
        g1[2] = 0x00010000;            // tensor_dim0 hi=0 | tensor_dim1 lo = 1
        g1[3] = (int)(1024u << 16);    // tensor_dim1 hi=0 | tile_dim0 = 1024
        g1[4] = 1;                     // tile_dim1 = 1, tile_dim2 = 0
        g1[5] = 1024;                  // tensor_dim0_stride lo32
        g1[6] = 0;
        g1[7] = 0;
        i32x4 z4; z4[0] = z4[1] = z4[2] = z4[3] = 0;
#if defined(__clang_major__) && __clang_major__ >= 23
        i32x8 z8;
        #pragma unroll
        for (int q = 0; q < 8; ++q) z8[q] = 0;
        __builtin_amdgcn_tensor_load_to_lds(g0, g1, z4, z4, z8, 0);
#else
        __builtin_amdgcn_tensor_load_to_lds(g0, g1, z4, z4, 0);
#endif
        __builtin_amdgcn_s_wait_tensorcnt(0);
    }
#else
    #pragma unroll
    for (int q = 0; q < 4; ++q)                     // cooperative fallback
        Wbuf[tid + 256 * q] = W[blkBase + tid + 256 * q];
#endif

    // mw2 -> registers (uniform address => scalar loads, no LDS traffic)
    float m2[32];
    #pragma unroll
    for (int h = 0; h < 32; ++h) m2[h] = mw2[h];

    // Constant B fragments for mw1^T (B[k,n]=mw1[n,k]; only K=0..2 nonzero,
    // which lives entirely in lanes 0-15 / elems 0..2 of the B layout).
    v16bf bfrag[2];
    #pragma unroll
    for (int ht = 0; ht < 2; ++ht) {
        #pragma unroll
        for (int e = 0; e < 16; ++e) bfrag[ht][e] = (__bf16)0.0f;
        if (hi == 0) {
            const int n = lr + 16 * ht;
            bfrag[ht][0] = (__bf16)mw1[n * 3 + 0];
            bfrag[ht][1] = (__bf16)mw1[n * 3 + 1];
            bfrag[ht][2] = (__bf16)mw1[n * 3 + 2];
        }
    }
    const float mb1v[2] = { mb1[lr], mb1[lr + 16] };
    const float mb2v = mb2[0];
    __syncthreads();                                // Wbuf ready for all waves

    #pragma unroll
    for (int it = 0; it < 4; ++it) {
        const unsigned loc  = (unsigned)it * 256u + wv * 32u;   // offset in Wbuf
        const unsigned base = blkBase + loc;                    // flat weight idx

        __builtin_prefetch(vin + ((base + 256u) & imask), 0, 0);

        #pragma unroll
        for (int mt = 0; mt < 2; ++mt) {
            // A tile: 16 triples in rows M=0..15; K=0,1,2 = (v_in, w, v_out).
            v16bf a;
            #pragma unroll
            for (int e = 0; e < 16; ++e) a[e] = (__bf16)0.0f;
            if (hi == 0) {
                const unsigned idx = base + mt * 16u + lr;
                a[0] = (__bf16)vin[idx & imask];
                a[1] = (__bf16)Wbuf[loc + mt * 16u + lr];
                a[2] = (__bf16)vout[idx >> ISHIFT];
            }
            #pragma unroll
            for (int ht = 0; ht < 2; ++ht) {
                v8f c;
                #pragma unroll
                for (int r = 0; r < 8; ++r) c[r] = mb1v[ht];   // + mb1 via C init
                c = __builtin_amdgcn_wmma_f32_16x16x32_bf16(
                        false, a, false, bfrag[ht], (short)0, c, false, false);
                #pragma unroll
                for (int r = 0; r < 8; ++r)                    // D -> (triple,hidden)
                    Hb[wv][r + 8 * hi + 16 * mt][lr + 16 * ht] = fmaxf(c[r], 0.0f);
            }
        }
        __syncthreads();

        // Layer 2: each lane owns one triple; dot(relu(H), mw2) + mb2.
        const float4* hrow = (const float4*)&Hb[wv][lane][0];  // ds_load_b128 x8
        float acc = mb2v;
        #pragma unroll
        for (int h4 = 0; h4 < 8; ++h4) {
            float4 hv = hrow[h4];
            acc = fmaf(hv.x, m2[4 * h4 + 0], acc);
            acc = fmaf(hv.y, m2[4 * h4 + 1], acc);
            acc = fmaf(hv.z, m2[4 * h4 + 2], acc);
            acc = fmaf(hv.w, m2[4 * h4 + 3], acc);
        }
        Wout[base + lane] = acc;          // coalesced 128B per wave
        __syncthreads();
    }
}

// ---------------------------------------------------------------------------
// Host-side launch. Inputs (setup_inputs order):
//   0:x 1:w1 2:b1 3:w2 4:b2 5:w3 6:b3 7:mw1 8:mb1 9:mw2 10:mb2
// d_out = [out(32x1024) | nw1(2048x1024) | nw2(2048x2048) | nw3(1024x2048)]
// d_ws holds a1, a2 fp32 activations (512 KB).
// ---------------------------------------------------------------------------
extern "C" void kernel_launch(void* const* d_in, const int* in_sizes, int n_in,
                              void* d_out, int out_size, void* d_ws, size_t ws_size,
                              hipStream_t stream) {
    (void)in_sizes; (void)n_in; (void)out_size; (void)ws_size;

    const float* x   = (const float*)d_in[0];
    const float* w1  = (const float*)d_in[1];
    const float* b1  = (const float*)d_in[2];
    const float* w2  = (const float*)d_in[3];
    const float* b2  = (const float*)d_in[4];
    const float* w3  = (const float*)d_in[5];
    const float* b3  = (const float*)d_in[6];
    const float* mw1 = (const float*)d_in[7];
    const float* mb1 = (const float*)d_in[8];
    const float* mw2 = (const float*)d_in[9];
    const float* mb2 = (const float*)d_in[10];

    const int I = 1024, H1 = 2048, H2 = 2048, O = 1024;

    float* a1  = (float*)d_ws;           // [32, 2048]
    float* a2  = a1 + 32 * H1;           // [32, 2048]
    float* out = (float*)d_out;          // [32, 1024]
    float* nw1 = out + 32 * O;           // [2048, 1024]
    float* nw2 = nw1 + (size_t)H1 * I;   // [2048, 2048]
    float* nw3 = nw2 + (size_t)H2 * H1;  // [1024, 2048]

    // Forward MLP (all on `stream`, in order)
    mlp_layer_kernel<1024, 2048><<<dim3(H1 / 128, 2), 256, 0, stream>>>(x,  w1, b1, a1);
    mlp_layer_kernel<2048, 2048><<<dim3(H2 / 128, 2), 256, 0, stream>>>(a1, w2, b2, a2);
    mlp_layer_kernel<2048, 1024><<<dim3(O  / 128, 2), 256, 0, stream>>>(a2, w3, b3, out);

    // Meta weight updates (each block = 1024 weights; totals divide evenly)
    meta_update_kernel<10><<<(H1 * I)  / 1024, 256, 0, stream>>>(w1, x,  a1,  mw1, mb1, mw2, mb2, nw1);
    meta_update_kernel<11><<<(H2 * H1) / 1024, 256, 0, stream>>>(w2, a1, a2,  mw1, mb1, mw2, mb2, nw2);
    meta_update_kernel<11><<<(O * H2)  / 1024, 256, 0, stream>>>(w3, a2, out, mw1, mb1, mw2, mb2, nw3);
}
